// TimesSeriesAttention_39874476376352
// MI455X (gfx1250) — matrix operand order repaired and checked by reading, not compiled, not run on hardware
//
#include <hip/hip_runtime.h>
#include <hip/hip_bf16.h>

typedef __attribute__((ext_vector_type(16))) __bf16 v16bf;
typedef __attribute__((ext_vector_type(8)))  __bf16 v8bf;
typedef __attribute__((ext_vector_type(8)))  float  v8f;

#define DEV static __device__ __forceinline__

DEV float bf2f(unsigned short u) {
    union { unsigned int i; float f; } x; x.i = ((unsigned int)u) << 16; return x.f;
}
DEV unsigned short f2bf(float f) {
    union { float f; unsigned int i; } x; x.f = f;
    unsigned int i = x.i;
    return (unsigned short)((i + 0x7FFFu + ((i >> 16) & 1u)) >> 16);
}
DEV v8f vzero() {
    v8f z;
#pragma unroll
    for (int r = 0; r < 8; ++r) z[r] = 0.f;
    return z;
}
// Assemble a 16-element bf16 fragment from two 16-byte chunks.
DEV v16bf make_frag(const unsigned short* p0, const unsigned short* p1) {
    v8bf lo = *(const v8bf*)p0;
    v8bf hi = *(const v8bf*)p1;
    return __builtin_shufflevector(lo, hi, 0,1,2,3,4,5,6,7,8,9,10,11,12,13,14,15);
}

// ---------------- prep kernels ----------------
__global__ void cast_bf16(const float* __restrict__ src, unsigned short* __restrict__ dst, int n) {
    int i = blockIdx.x * blockDim.x + threadIdx.x;
    int stride = gridDim.x * blockDim.x;
    for (; i < n; i += stride) dst[i] = f2bf(src[i]);
}
// src: rows x cols (row-major). dst: rows x cols (N-major, K contiguous for B-operand loads).
__global__ void transpose_cast(const float* __restrict__ src, unsigned short* __restrict__ dst,
                               int rows, int cols) {
    int i = blockIdx.x * blockDim.x + threadIdx.x;
    int total = rows * cols;
    int stride = gridDim.x * blockDim.x;
    for (; i < total; i += stride) {
        int n = i % rows;
        int k = i / rows;
        dst[(size_t)n * cols + k] = f2bf(src[(size_t)n * cols + k]);
    }
}
// 21x64 f32 -> 64x32 bf16 [d][tap] zero-padded (K padded to 32 for WMMA).
__global__ void pad_cast(const float* __restrict__ src, unsigned short* __restrict__ dst) {
    int i = blockIdx.x * blockDim.x + threadIdx.x;
    if (i < 32 * 64) {
        int tap = i >> 6, d = i & 63;
        dst[d * 32 + tap] = (tap < 21) ? f2bf(src[tap * 64 + d]) : (unsigned short)0;
    }
}

// ---------------- GEMM1: Xp1 = silu(Xb @ W1bT), bf16 out ----------------
// grid: (N/256, 640/64), block 256 (8 waves). Wave -> 32x64 tile, K=256.
// Double-buffered K-steps: next step's fragments are in flight while the
// current step's 8 WMMAs execute (enables partial s_wait_loadcnt).
__global__ __launch_bounds__(256)
void gemm1_silu(const unsigned short* __restrict__ Xb,
                const unsigned short* __restrict__ W1bT,
                unsigned short* __restrict__ Xp1) {
    const int lane = threadIdx.x & 31;
    const int wave = threadIdx.x >> 5;
    const int mtile = (blockIdx.x * 8 + wave) * 32;
    const int ntile = blockIdx.y * 64;
    const int m = lane & 15;
    const int hi = (lane >> 4) & 1;

    v8f acc[2][4];
#pragma unroll
    for (int mi = 0; mi < 2; ++mi)
#pragma unroll
        for (int t = 0; t < 4; ++t) acc[mi][t] = vzero();

    const unsigned short* arow0 = Xb + (size_t)(mtile + m) * 256 + hi * 8;
    const unsigned short* arow1 = arow0 + 16 * 256;

    // prologue: K-step 0
    v16bf a0 = make_frag(arow0, arow0 + 16);
    v16bf a1 = make_frag(arow1, arow1 + 16);
    v16bf b[4];
    {
        const unsigned short* brow = W1bT + (size_t)(ntile + m) * 256 + hi * 8;
#pragma unroll
        for (int t = 0; t < 4; ++t) b[t] = make_frag(brow + 16 * t * 256, brow + 16 * t * 256 + 16);
    }

#pragma unroll
    for (int kk = 0; kk < 256; kk += 32) {
        v16bf na0, na1, nb[4];
        if (kk + 32 < 256) {  // prefetch next K-step
            na0 = make_frag(arow0 + kk + 32, arow0 + kk + 48);
            na1 = make_frag(arow1 + kk + 32, arow1 + kk + 48);
            const unsigned short* brow = W1bT + (size_t)(ntile + m) * 256 + kk + 32 + hi * 8;
#pragma unroll
            for (int t = 0; t < 4; ++t) nb[t] = make_frag(brow + 16 * t * 256, brow + 16 * t * 256 + 16);
        }
#pragma unroll
        for (int t = 0; t < 4; ++t) {
            acc[0][t] = __builtin_amdgcn_wmma_f32_16x16x32_bf16(
                false, a0, false, b[t], (short)0, acc[0][t], false, false);
            acc[1][t] = __builtin_amdgcn_wmma_f32_16x16x32_bf16(
                false, a1, false, b[t], (short)0, acc[1][t], false, false);
        }
        a0 = na0; a1 = na1;
#pragma unroll
        for (int t = 0; t < 4; ++t) b[t] = nb[t];
    }

#pragma unroll
    for (int mi = 0; mi < 2; ++mi)
#pragma unroll
        for (int t = 0; t < 4; ++t)
#pragma unroll
            for (int r = 0; r < 8; ++r) {
                float x = acc[mi][t][r];
                float s = x / (1.f + __expf(-x));   // silu
                Xp1[(size_t)(mtile + mi * 16 + r + 8 * hi) * 640 + ntile + 16 * t + (lane & 15)] =
                    f2bf(s);
            }
}

// ---------------- GEMM2 + fused per-64-chunk l2norm ----------------
// Xn = l2norm_64(Xp1 @ W2bT), bf16 out. Wave's 64 columns == one norm chunk.
__global__ __launch_bounds__(256)
void gemm2_norm(const unsigned short* __restrict__ Xp1,
                const unsigned short* __restrict__ W2bT,
                unsigned short* __restrict__ Xn) {
    const int lane = threadIdx.x & 31;
    const int wave = threadIdx.x >> 5;
    const int mtile = (blockIdx.x * 8 + wave) * 32;
    const int ntile = blockIdx.y * 64;
    const int m = lane & 15;
    const int hi = (lane >> 4) & 1;

    v8f acc[2][4];
#pragma unroll
    for (int mi = 0; mi < 2; ++mi)
#pragma unroll
        for (int t = 0; t < 4; ++t) acc[mi][t] = vzero();

    const unsigned short* arow0 = Xp1 + (size_t)(mtile + m) * 640 + hi * 8;
    const unsigned short* arow1 = arow0 + 16 * 640;

    v16bf a0 = make_frag(arow0, arow0 + 16);
    v16bf a1 = make_frag(arow1, arow1 + 16);
    v16bf b[4];
    {
        const unsigned short* brow = W2bT + (size_t)(ntile + m) * 640 + hi * 8;
#pragma unroll
        for (int t = 0; t < 4; ++t) b[t] = make_frag(brow + 16 * t * 640, brow + 16 * t * 640 + 16);
    }

#pragma unroll
    for (int kk = 0; kk < 640; kk += 32) {
        v16bf na0, na1, nb[4];
        if (kk + 32 < 640) {
            na0 = make_frag(arow0 + kk + 32, arow0 + kk + 48);
            na1 = make_frag(arow1 + kk + 32, arow1 + kk + 48);
            const unsigned short* brow = W2bT + (size_t)(ntile + m) * 640 + kk + 32 + hi * 8;
#pragma unroll
            for (int t = 0; t < 4; ++t) nb[t] = make_frag(brow + 16 * t * 640, brow + 16 * t * 640 + 16);
        }
#pragma unroll
        for (int t = 0; t < 4; ++t) {
            acc[0][t] = __builtin_amdgcn_wmma_f32_16x16x32_bf16(
                false, a0, false, b[t], (short)0, acc[0][t], false, false);
            acc[1][t] = __builtin_amdgcn_wmma_f32_16x16x32_bf16(
                false, a1, false, b[t], (short)0, acc[1][t], false, false);
        }
        a0 = na0; a1 = na1;
#pragma unroll
        for (int t = 0; t < 4; ++t) b[t] = nb[t];
    }

    // Row's 64 values live on 16 lanes (one half-wave) x 4 accum tiles.
#pragma unroll
    for (int mi = 0; mi < 2; ++mi)
#pragma unroll
        for (int r = 0; r < 8; ++r) {
            float s = 0.f;
#pragma unroll
            for (int t = 0; t < 4; ++t) s += acc[mi][t][r] * acc[mi][t][r];
            s += __shfl_xor(s, 1, 32);
            s += __shfl_xor(s, 2, 32);
            s += __shfl_xor(s, 4, 32);
            s += __shfl_xor(s, 8, 32);
            float inv = rsqrtf(s + 1e-6f);
#pragma unroll
            for (int t = 0; t < 4; ++t)
                Xn[(size_t)(mtile + mi * 16 + r + 8 * hi) * 640 + ntile + 16 * t + (lane & 15)] =
                    f2bf(acc[mi][t][r] * inv);
        }
}

// ---------------- Stage C: per-token patch GEMMs + grouped conv ----------------
// One wave (32 threads) per token. LDS ~34KB.
__global__ __launch_bounds__(32)
void token_kernel(const unsigned short* __restrict__ Xn,
                  const float* __restrict__ P1,
                  const unsigned short* __restrict__ H1wb,  // [tap][c][d] bf16
                  const unsigned short* __restrict__ H2pad, // [64][32] bf16 [d][tap]
                  const unsigned short* __restrict__ Gpad,  // [64][32] bf16 [d][tap]
                  float* __restrict__ out) {
    __shared__ __align__(16) unsigned short sVp[64 * 32];  // vpatch^T, K-padded to 32
    __shared__ __align__(16) unsigned short sKp[21 * 64];  // kpatch [tap][c]
    __shared__ __align__(16) unsigned short sQ[8 * 64];    // q heads
    __shared__ __align__(16) unsigned short sV[64 * 64];   // V^T [d][v]
    __shared__ __align__(16) unsigned short sVGT[64 * 64]; // VG^T [c][v]
    __shared__ __align__(16) unsigned short sFeat[64 * 64];
    __shared__ __align__(16) float sKer[8 * 64];

    const int lane = threadIdx.x;
    const int tok = blockIdx.x;
    const int h = tok & 31;
    const int ti = (tok >> 5) & 63;
    const int mlane = lane & 15;
    const int hi = lane >> 4;

    // zero pad regions
    for (int i = lane; i < 64 * 32 / 2; i += 32) ((unsigned int*)sVp)[i] = 0u;
    for (int i = lane; i < 21 * 32; i += 32) ((unsigned int*)sKp)[i] = 0u;
    {   // stage q (8x64 bf16)
        const unsigned int* src = (const unsigned int*)(Xn + (size_t)tok * 640);
        for (int i = lane; i < 256; i += 32) ((unsigned int*)sQ)[i] = src[i];
    }
    __syncthreads();

    // gather 3x7 patches of k and v (zero-padded borders)
    const int c0 = lane * 2;
    for (int tap = 0; tap < 21; ++tap) {
        int di = tap / 7 - 1, dj = tap % 7 - 3;
        int t2 = ti + di, h2 = h + dj;
        unsigned int kv = 0u, vv = 0u;
        if ((unsigned)t2 < 64u && (unsigned)h2 < 32u) {
            const unsigned short* srow = Xn + (size_t)(tok + di * 32 + dj) * 640;
            kv = *(const unsigned int*)(srow + 512 + c0);
            vv = *(const unsigned int*)(srow + 576 + c0);
        }
        ((unsigned int*)sKp)[tap * 32 + lane] = kv;
        sVp[c0 * 32 + tap]       = (unsigned short)(vv & 0xFFFFu);
        sVp[(c0 + 1) * 32 + tap] = (unsigned short)(vv >> 16);
    }
    __syncthreads();

    // Phase 1: V = vpatch^T @ H2w ; VG = vpatch^T @ Gw + I   (WMMA, K=32)
    v16bf bH2[4], bG[4];
    {
        const unsigned short* h2r = H2pad + mlane * 32 + hi * 8;
        const unsigned short* gr  = Gpad + mlane * 32 + hi * 8;
#pragma unroll
        for (int tn = 0; tn < 4; ++tn) {
            bH2[tn] = make_frag(h2r + 16 * tn * 32, h2r + 16 * tn * 32 + 16);
            bG[tn]  = make_frag(gr + 16 * tn * 32, gr + 16 * tn * 32 + 16);
        }
    }
#pragma unroll
    for (int tm = 0; tm < 4; ++tm) {
        const unsigned short* ap = sVp + (16 * tm + mlane) * 32 + hi * 8;
        v16bf a = make_frag(ap, ap + 16);
#pragma unroll
        for (int tn = 0; tn < 4; ++tn) {
            v8f accV = __builtin_amdgcn_wmma_f32_16x16x32_bf16(
                false, a, false, bH2[tn], (short)0, vzero(), false, false);
            v8f accG = __builtin_amdgcn_wmma_f32_16x16x32_bf16(
                false, a, false, bG[tn], (short)0, vzero(), false, false);
#pragma unroll
            for (int r = 0; r < 8; ++r) {
                int mm = 16 * tm + r + 8 * hi;
                int nn = 16 * tn + mlane;
                sV[nn * 64 + mm] = f2bf(accV[r]);
                float g = accG[r] + ((mm == nn) ? 1.f : 0.f);
                sVGT[nn * 64 + mm] = f2bf(g); // store transposed for feature A-operand
            }
        }
    }
    __syncthreads();

    // Phase 2: K_H1 (grouped conv, per-channel weights -> VALU) fused into kernel = q @ K_H1
    float kern[8][2];
#pragma unroll
    for (int n = 0; n < 8; ++n) { kern[n][0] = 0.f; kern[n][1] = 0.f; }
    const int d0 = lane * 2;
    for (int c = 0; c < 64; ++c) {
        float kh0 = P1[c * 64 + d0];
        float kh1 = P1[c * 64 + d0 + 1];
#pragma unroll
        for (int tap = 0; tap < 21; ++tap) {
            float kc = bf2f(sKp[tap * 64 + c]); // LDS broadcast
            unsigned int w2 = *(const unsigned int*)(H1wb + (size_t)tap * 4096 + c * 64 + d0);
            kh0 += kc * bf2f((unsigned short)(w2 & 0xFFFFu));
            kh1 += kc * bf2f((unsigned short)(w2 >> 16));
        }
#pragma unroll
        for (int n = 0; n < 8; ++n) {
            float qv = bf2f(sQ[n * 64 + c]);
            kern[n][0] += qv * kh0;
            kern[n][1] += qv * kh1;
        }
    }
#pragma unroll
    for (int n = 0; n < 8; ++n) {
        sKer[n * 64 + d0] = kern[n][0];
        sKer[n * 64 + d0 + 1] = kern[n][1];
    }
    __syncthreads();

    // Phase 3: feature = VG^T @ V   (WMMA, 64x64x64)
    v16bf bF[4][2];
#pragma unroll
    for (int tn = 0; tn < 4; ++tn)
#pragma unroll
        for (int kh = 0; kh < 2; ++kh) {
            const unsigned short* p = sV + (16 * tn + mlane) * 64 + kh * 32 + hi * 8;
            bF[tn][kh] = make_frag(p, p + 16);
        }
#pragma unroll
    for (int tm = 0; tm < 4; ++tm) {
        const unsigned short* ap = sVGT + (16 * tm + mlane) * 64 + hi * 8;
        v16bf a0 = make_frag(ap, ap + 16);
        v16bf a1 = make_frag(ap + 32, ap + 48);
#pragma unroll
        for (int tn = 0; tn < 4; ++tn) {
            v8f acc = vzero();
            acc = __builtin_amdgcn_wmma_f32_16x16x32_bf16(
                false, a0, false, bF[tn][0], (short)0, acc, false, false);
            acc = __builtin_amdgcn_wmma_f32_16x16x32_bf16(
                false, a1, false, bF[tn][1], (short)0, acc, false, false);
#pragma unroll
            for (int r = 0; r < 8; ++r)
                sFeat[(16 * tm + r + 8 * hi) * 64 + 16 * tn + mlane] = f2bf(acc[r]);
        }
    }
    __syncthreads();

    // Phase 4: out[v,n] = sum_d feature[v,d]*kernel[n,d]  (small, VALU)
    float o[8][2];
#pragma unroll
    for (int n = 0; n < 8; ++n) { o[n][0] = 0.f; o[n][1] = 0.f; }
    for (int d = 0; d < 64; ++d) {
        float f0 = bf2f(sFeat[lane * 64 + d]);
        float f1 = bf2f(sFeat[(lane + 32) * 64 + d]);
#pragma unroll
        for (int n = 0; n < 8; ++n) {
            float kr = sKer[n * 64 + d];
            o[n][0] += kr * f0;
            o[n][1] += kr * f1;
        }
    }
    float* dst = out + (size_t)tok * 512;
#pragma unroll
    for (int n = 0; n < 8; ++n) {
        dst[lane * 8 + n] = o[n][0];
        dst[(lane + 32) * 8 + n] = o[n][1];
    }
}

// ---------------- host ----------------
extern "C" void kernel_launch(void* const* d_in, const int* in_sizes, int n_in,
                              void* d_out, int out_size, void* d_ws, size_t ws_size,
                              hipStream_t stream) {
    const float* x   = (const float*)d_in[0];
    const float* W1  = (const float*)d_in[1];
    const float* W2  = (const float*)d_in[2];
    const float* P1  = (const float*)d_in[3];
    const float* H1w = (const float*)d_in[4];
    const float* H2w = (const float*)d_in[5];
    const float* Gw  = (const float*)d_in[6];
    // d_in[7] = identity, folded in-kernel

    const int N = 8 * 64 * 32; // 16384 tokens
    char* ws = (char*)d_ws;
    size_t off = 0;
    auto carve = [&](size_t bytes) -> char* {
        char* p = ws + off;
        off = (off + bytes + 255) & ~(size_t)255;
        return p;
    };
    unsigned short* Xb   = (unsigned short*)carve((size_t)N * 256 * 2);
    unsigned short* W1bT = (unsigned short*)carve((size_t)256 * 640 * 2);
    unsigned short* W2bT = (unsigned short*)carve((size_t)640 * 640 * 2);
    unsigned short* Xp1  = (unsigned short*)carve((size_t)N * 640 * 2);
    unsigned short* XnB  = (unsigned short*)carve((size_t)N * 640 * 2);
    unsigned short* H1wb = (unsigned short*)carve((size_t)21 * 64 * 64 * 2);
    unsigned short* H2p  = (unsigned short*)carve((size_t)32 * 64 * 2);
    unsigned short* Gp   = (unsigned short*)carve((size_t)32 * 64 * 2);

    cast_bf16<<<2048, 256, 0, stream>>>(x, Xb, N * 256);
    transpose_cast<<<640, 256, 0, stream>>>(W1, W1bT, 640, 256);
    transpose_cast<<<1600, 256, 0, stream>>>(W2, W2bT, 640, 640);
    cast_bf16<<<336, 256, 0, stream>>>(H1w, H1wb, 21 * 64 * 64);
    pad_cast<<<8, 256, 0, stream>>>(H2w, H2p);
    pad_cast<<<8, 256, 0, stream>>>(Gw, Gp);

    dim3 gg(N / 256, 10);
    gemm1_silu<<<gg, 256, 0, stream>>>(Xb, W1bT, Xp1);
    gemm2_norm<<<gg, 256, 0, stream>>>(Xp1, W2bT, XnB);
    token_kernel<<<N, 32, 0, stream>>>(XnB, P1, H1wb, H2p, Gp, (float*)d_out);
}
